// LSTMEncoder_23922967839087
// MI455X (gfx1250) — matrix operand order repaired and checked by reading, not compiled, not run on hardware
//
#include <hip/hip_runtime.h>
#include <cstdint>

// ---------------------------------------------------------------------------
// LSTMEncoder for MI455X (gfx1250).
//   conv1 (Cin=9):           fp32 VALU -> f16
//   conv2/conv3:             implicit-GEMM f16 WMMA 16x16x32, BN+ReLU fused
//   reversed-time LSTM:      f16 WMMA, weights resident in VGPRs, x_t staged
//                            via GLOBAL_LOAD_ASYNC_TO_LDS_B128 (double buffer),
//                            A-fragments pre-loaded so the 16-WMMA chain runs
//                            behind a single s_wait_dscnt per step.
// ---------------------------------------------------------------------------

#define EPS 1e-5f

typedef __attribute__((ext_vector_type(16))) _Float16 v16h;
typedef __attribute__((ext_vector_type(8)))  float    v8f;

constexpr int Bb  = 64;     // batch
constexpr int Tt  = 4096;   // time
constexpr int TSH = 12;     // log2(Tt)
constexpr int Hh  = 128;    // hidden = feature
constexpr int G4  = 512;    // 4*H gate columns
constexpr int KK  = 7;      // conv kernel width

// ----------------------------- WMMA fragments ------------------------------
// A fragment (16x32 f16, ISA 7.12.2): lane m = lane&15;
// K interleave: VGPR0..3 = K{0..7|8..15}, VGPR4..7 = K{16..23|24..31}.
__device__ __forceinline__ v16h load_a_frag(const _Float16* __restrict__ A,
                                            int ld, int kt, int lane) {
  int mrow = lane & 15;
  int kb   = kt * 32 + ((lane >> 4) << 3);
  const _Float16* p = A + mrow * ld + kb;
  v16h r;
#pragma unroll
  for (int i = 0; i < 8; ++i) { r[i] = p[i]; r[8 + i] = p[16 + i]; }
  return r;
}

// B fragment (32x16 f16 tile of W^T): lane n = lane&15; same K interleave as
// A: elems 0..7 = K{0..7|8..15}, elems 8..15 = K{16..23|24..31}.
__device__ __forceinline__ v16h load_b_frag(const _Float16* __restrict__ Wrow,
                                            int kt, int lane) {
  const _Float16* p = Wrow + kt * 32 + ((lane >> 4) << 3);
  v16h r;
#pragma unroll
  for (int i = 0; i < 8; ++i) { r[i] = p[i]; r[8 + i] = p[16 + i]; }
  return r;
}

__device__ __forceinline__ float sigf(float x) {
  return __builtin_amdgcn_rcpf(1.f + __expf(-x));
}
__device__ __forceinline__ float tanh_fast(float x) {
  return 1.f - 2.f * __builtin_amdgcn_rcpf(__expf(2.f * x) + 1.f);
}

// ------------------------- conv1: fp32 VALU -> f16 -------------------------
template <int CI, int CO>
__global__ void convbn_valu_kernel(const float* __restrict__ in,
                                   const float* __restrict__ w,
                                   const float* __restrict__ g,
                                   const float* __restrict__ bb,
                                   const float* __restrict__ m,
                                   const float* __restrict__ v,
                                   _Float16* __restrict__ out) {
  int idx = blockIdx.x * blockDim.x + threadIdx.x;
  if (idx >= Bb * Tt * CO) return;
  int co  = idx % CO;
  int rem = idx / CO;
  int t   = rem & (Tt - 1);
  int b   = rem >> TSH;

  const float* wr = w + co * CI * KK;
  float acc = 0.f;
#pragma unroll
  for (int k = 0; k < KK; ++k) {
    int tt = t + k - (KK / 2);
    if ((unsigned)tt < (unsigned)Tt) {
      const float* xr = in + ((size_t)b * Tt + tt) * CI;
#pragma unroll
      for (int ci = 0; ci < CI; ++ci) acc += xr[ci] * wr[ci * KK + k];
    }
  }
  float sc = g[co] * rsqrtf(v[co] + EPS);
  float y  = (acc - m[co]) * sc + bb[co];
  out[idx] = (_Float16)fmaxf(y, 0.f);
}

// -------------------- weight prep: [CO,CI,K]f32 -> [K,CO,CI]f16 ------------
template <int CI, int CO>
__global__ void prep_convw_kernel(const float* __restrict__ w,
                                  _Float16* __restrict__ Wk) {
  int i = blockIdx.x * blockDim.x + threadIdx.x;
  if (i >= KK * CO * CI) return;
  int ci = i % CI;
  int co = (i / CI) % CO;
  int k  = i / (CI * CO);
  Wk[i] = (_Float16)w[(co * CI + ci) * KK + k];
}

__global__ void prep_lstm_kernel(const float* __restrict__ wih,
                                 const float* __restrict__ whh,
                                 const float* __restrict__ bih,
                                 const float* __restrict__ bhh,
                                 _Float16* __restrict__ Wih,
                                 _Float16* __restrict__ Whh,
                                 float* __restrict__ bias) {
  int i = blockIdx.x * blockDim.x + threadIdx.x;
  if (i < G4 * Hh) {
    Wih[i] = (_Float16)wih[i];
    Whh[i] = (_Float16)whh[i];
  }
  if (i < G4) bias[i] = bih[i] + bhh[i];
}

// --------------- conv2/conv3: implicit-GEMM WMMA, BN+ReLU fused ------------
// out[b,t,co] = relu(bn(sum_k sum_ci in[b,t+k-3,ci] * w[co,ci,k]))
// Workgroup: one b, 64 t-rows (4 M-tiles) x all CO columns (CO/16 N-tiles).
// Each wave owns one fixed N-tile column; weight B-fragments stay in VGPRs.
template <int CI, int CO>
__global__ __launch_bounds__(512) void conv_wmma_kernel(
    const _Float16* __restrict__ in,   // [B, T, CI] f16
    const _Float16* __restrict__ Wk,   // [K][CO][CI] f16
    const float* __restrict__ g, const float* __restrict__ bb,
    const float* __restrict__ m, const float* __restrict__ v,
    _Float16* __restrict__ out) {      // [B, T, CO] f16
  constexpr int NT    = CO / 16;       // N tiles
  constexpr int KT    = CI / 32;       // K frags per tap
  constexpr int MSTEP = 16 / NT;       // M-tile stride per wave
  constexpr int ROWS  = 64 + KK - 1;   // staged rows incl. halo

  __shared__ __align__(16) _Float16 xs[ROWS * CI];

  const int tid  = threadIdx.x;
  const int lane = tid & 31;
  const int wv   = tid >> 5;
  const int b    = blockIdx.x / (Tt / 64);
  const int t0   = (blockIdx.x % (Tt / 64)) * 64;

  // stage [t0-3, t0+66] x CI with zero halo
  for (int i = tid; i < ROWS * CI / 2; i += 512) {
    int r  = i / (CI / 2);
    int c  = (i - r * (CI / 2)) * 2;
    int tr = t0 + r - (KK / 2);
    uint32_t val = 0;
    if ((unsigned)tr < (unsigned)Tt)
      val = *(const uint32_t*)&in[((size_t)b * Tt + tr) * CI + c];
    *(uint32_t*)&xs[r * CI + c] = val;
  }

  const int n  = wv % NT;              // fixed N-tile for this wave
  const int m0 = wv / NT;

  // loop-invariant weight fragments
  v16h wb[KK][KT];
#pragma unroll
  for (int k = 0; k < KK; ++k)
#pragma unroll
    for (int kt = 0; kt < KT; ++kt)
      wb[k][kt] = load_b_frag(
          Wk + ((size_t)k * CO + n * 16 + (lane & 15)) * CI, kt, lane);

  const int   ncol  = n * 16 + (lane & 15);
  const int   mbase = (lane >> 4) * 8;
  const float sc    = g[ncol] * rsqrtf(v[ncol] + EPS);
  const float sh    = bb[ncol] - m[ncol] * sc;

  __syncthreads();

  for (int mt = m0; mt < 4; mt += MSTEP) {
    v8f acc = {0.f, 0.f, 0.f, 0.f, 0.f, 0.f, 0.f, 0.f};
#pragma unroll
    for (int k = 0; k < KK; ++k) {
      const _Float16* arow = xs + (mt * 16 + k) * CI;
#pragma unroll
      for (int kt = 0; kt < KT; ++kt) {
        v16h a = load_a_frag(arow, CI, kt, lane);
        acc = __builtin_amdgcn_wmma_f32_16x16x32_f16(
            false, a, false, wb[k][kt], (short)0, acc, false, false);
      }
    }
#pragma unroll
    for (int r = 0; r < 8; ++r) {
      int   trow = t0 + mt * 16 + mbase + r;
      float y    = fmaxf(acc[r] * sc + sh, 0.f);
      out[((size_t)b * Tt + trow) * CO + ncol] = (_Float16)y;
    }
  }
}

// ------------------------------- LSTM kernel -------------------------------
// 4 blocks x 16 batch rows, 512 threads = 16 waves; each wave owns 2 N-tiles
// (16 weight B-fragments pinned in VGPRs for all 4096 steps). x_t staged via
// async global->LDS B128 copies, double-buffered to overlap the GEMM. All 8
// A-fragments are pre-loaded into registers so the LDS waits collapse to one.
__global__ __launch_bounds__(512) void lstm_kernel(
    const _Float16* __restrict__ feat,   // [B, T, H] f16
    const _Float16* __restrict__ Wih,    // [4H, H] f16
    const _Float16* __restrict__ Whh,    // [4H, H] f16
    const float* __restrict__ bias,      // [4H] = b_ih + b_hh
    float* __restrict__ out) {           // [B, T, H] f32
  __shared__ __align__(16) _Float16 xbuf[2][16 * Hh];
  __shared__ __align__(16) _Float16 hbuf[16 * Hh];
  __shared__ float gbuf[16 * G4];
  __shared__ float sbias[G4];

  const int tid  = threadIdx.x;
  const int lane = tid & 31;
  const int wv   = tid >> 5;
  const int b0   = blockIdx.x * 16;
  const uint32_t xbase = (uint32_t)(uintptr_t)&xbuf[0][0];

  // loop-invariant LSTM weight fragments -> registers
  v16h fih[2][4], fhh[2][4];
#pragma unroll
  for (int n2 = 0; n2 < 2; ++n2)
#pragma unroll
    for (int kt = 0; kt < 4; ++kt) {
      fih[n2][kt] =
          load_b_frag(Wih + (size_t)((wv * 2 + n2) * 16 + (lane & 15)) * Hh,
                      kt, lane);
      fhh[n2][kt] =
          load_b_frag(Whh + (size_t)((wv * 2 + n2) * 16 + (lane & 15)) * Hh,
                      kt, lane);
    }

  for (int i = tid; i < 16 * Hh; i += 512) hbuf[i] = (_Float16)0.f;
  for (int i = tid; i < G4; i += 512) sbias[i] = bias[i];
  float creg[4] = {0.f, 0.f, 0.f, 0.f};
  const int idx0 = tid * 4;

  // prefetch x_{T-1} into buffer 0 (16 rows x 256B, one B128 per lane)
  if (tid < 256) {
    int row = tid >> 4, seg = tid & 15;
    const _Float16* src =
        feat + ((size_t)(b0 + row) * Tt + (Tt - 1)) * Hh + seg * 8;
    uint32_t dst = xbase + row * (Hh * 2) + seg * 16;
    asm volatile("global_load_async_to_lds_b128 %0, %1, off"
                 :: "v"(dst), "v"(src) : "memory");
  }

  int cur = 0;
  for (int t = Tt - 1; t >= 0; --t) {
    asm volatile("s_wait_asynccnt 0x0" ::: "memory");
    __syncthreads();   // x_t landed; previous activation's hbuf published

    // kick off async copy of x_{t-1} into the other buffer (overlaps GEMM)
    if (t > 0 && tid < 256) {
      int row = tid >> 4, seg = tid & 15;
      const _Float16* src =
          feat + ((size_t)(b0 + row) * Tt + (t - 1)) * Hh + seg * 8;
      uint32_t dst =
          xbase + (cur ^ 1) * (16 * Hh * 2) + row * (Hh * 2) + seg * 16;
      asm volatile("global_load_async_to_lds_b128 %0, %1, off"
                   :: "v"(dst), "v"(src) : "memory");
    }

    // pre-load all 8 A-fragments -> loads pipeline, single dscnt wait
    const _Float16* xcur = &xbuf[cur][0];
    v16h ax[4], ah[4];
#pragma unroll
    for (int kt = 0; kt < 4; ++kt) {
      ax[kt] = load_a_frag(xcur, Hh, kt, lane);
      ah[kt] = load_a_frag(hbuf, Hh, kt, lane);
    }

#pragma unroll
    for (int n2 = 0; n2 < 2; ++n2) {
      v8f acc = {0.f, 0.f, 0.f, 0.f, 0.f, 0.f, 0.f, 0.f};
#pragma unroll
      for (int kt = 0; kt < 4; ++kt)
        acc = __builtin_amdgcn_wmma_f32_16x16x32_f16(
            false, ax[kt], false, fih[n2][kt], (short)0, acc, false, false);
#pragma unroll
      for (int kt = 0; kt < 4; ++kt)
        acc = __builtin_amdgcn_wmma_f32_16x16x32_f16(
            false, ah[kt], false, fhh[n2][kt], (short)0, acc, false, false);
      int ncol  = (wv * 2 + n2) * 16 + (lane & 15);
      int mbase = (lane >> 4) * 8;
#pragma unroll
      for (int r = 0; r < 8; ++r) gbuf[(mbase + r) * G4 + ncol] = acc[r];
    }
    __syncthreads();

    // activations: c = sig(f)*c + sig(i)*tanh(g); h = sig(o)*tanh(c)
#pragma unroll
    for (int p = 0; p < 4; ++p) {
      int idx = idx0 + p;
      int br  = idx >> 7;
      int j   = idx & 127;
      const float* gr = gbuf + br * G4;
      float ig = gr[j]       + sbias[j];
      float fg = gr[128 + j] + sbias[128 + j];
      float gg = gr[256 + j] + sbias[256 + j];
      float og = gr[384 + j] + sbias[384 + j];
      float c  = sigf(fg) * creg[p] + sigf(ig) * tanh_fast(gg);
      float h  = sigf(og) * tanh_fast(c);
      creg[p]  = c;
      hbuf[br * Hh + j] = (_Float16)h;
      out[((size_t)(b0 + br) * Tt + t) * Hh + j] = h;
    }
    cur ^= 1;
    // loop-top barrier orders hbuf writes before the next GEMM reads them
  }
}

// ------------------------------- launcher ----------------------------------
extern "C" void kernel_launch(void* const* d_in, const int* in_sizes, int n_in,
                              void* d_out, int out_size, void* d_ws,
                              size_t ws_size, hipStream_t stream) {
  const float* x    = (const float*)d_in[0];
  const float* c1w  = (const float*)d_in[1];
  const float* bn1g = (const float*)d_in[2];
  const float* bn1b = (const float*)d_in[3];
  const float* bn1m = (const float*)d_in[4];
  const float* bn1v = (const float*)d_in[5];
  const float* c2w  = (const float*)d_in[6];
  const float* bn2g = (const float*)d_in[7];
  const float* bn2b = (const float*)d_in[8];
  const float* bn2m = (const float*)d_in[9];
  const float* bn2v = (const float*)d_in[10];
  const float* c3w  = (const float*)d_in[11];
  const float* bn3g = (const float*)d_in[12];
  const float* bn3b = (const float*)d_in[13];
  const float* bn3m = (const float*)d_in[14];
  const float* bn3v = (const float*)d_in[15];
  const float* wih  = (const float*)d_in[16];
  const float* whh  = (const float*)d_in[17];
  const float* bih  = (const float*)d_in[18];
  const float* bhh  = (const float*)d_in[19];

  char* ws = (char*)d_ws;
  size_t off = 0;
  auto alloc = [&](size_t bytes) -> void* {
    void* p = ws + off;
    off += (bytes + 255) & ~(size_t)255;
    return p;
  };
  _Float16* o1   = (_Float16*)alloc((size_t)Bb * Tt * 32 * sizeof(_Float16));
  _Float16* o2   = (_Float16*)alloc((size_t)Bb * Tt * 64 * sizeof(_Float16));
  _Float16* feat = (_Float16*)alloc((size_t)Bb * Tt * Hh * sizeof(_Float16));
  _Float16* Wk2  = (_Float16*)alloc((size_t)KK * 64 * 32 * sizeof(_Float16));
  _Float16* Wk3  = (_Float16*)alloc((size_t)KK * 128 * 64 * sizeof(_Float16));
  _Float16* Wih  = (_Float16*)alloc((size_t)G4 * Hh * sizeof(_Float16));
  _Float16* Whh  = (_Float16*)alloc((size_t)G4 * Hh * sizeof(_Float16));
  float*    bias = (float*)alloc((size_t)G4 * sizeof(float));

  const int blk = 256;

  // weight prep
  int nw2 = KK * 64 * 32;
  prep_convw_kernel<32, 64>
      <<<(nw2 + blk - 1) / blk, blk, 0, stream>>>(c2w, Wk2);
  int nw3 = KK * 128 * 64;
  prep_convw_kernel<64, 128>
      <<<(nw3 + blk - 1) / blk, blk, 0, stream>>>(c3w, Wk3);
  int np = G4 * Hh;
  prep_lstm_kernel<<<(np + blk - 1) / blk, blk, 0, stream>>>(
      wih, whh, bih, bhh, Wih, Whh, bias);

  // conv stack
  int n1 = Bb * Tt * 32;
  convbn_valu_kernel<9, 32>
      <<<(n1 + blk - 1) / blk, blk, 0, stream>>>(x, c1w, bn1g, bn1b, bn1m,
                                                 bn1v, o1);
  conv_wmma_kernel<32, 64>
      <<<Bb * (Tt / 64), 512, 0, stream>>>(o1, Wk2, bn2g, bn2b, bn2m, bn2v,
                                           o2);
  conv_wmma_kernel<64, 128>
      <<<Bb * (Tt / 64), 512, 0, stream>>>(o2, Wk3, bn3g, bn3b, bn3m, bn3v,
                                           feat);

  // reversed-time LSTM
  lstm_kernel<<<Bb / 16, 512, 0, stream>>>(feat, Wih, Whh, bias,
                                           (float*)d_out);
}
